// EdgeMPNN_17806934409783
// MI455X (gfx1250) — matrix operand order repaired and checked by reading, not compiled, not run on hardware
//
#include <hip/hip_runtime.h>
#include <hip/hip_bf16.h>

#define NN 50000
#define NE 400000
#define EPS 1e-5f

typedef __bf16 bf16_t;
typedef __attribute__((ext_vector_type(16))) __bf16        v16bf;
typedef __attribute__((ext_vector_type(8)))  float         v8f;
typedef __attribute__((ext_vector_type(8)))  unsigned int  v8u;

// ---------------------------------------------------------------------------
// WMMA gather-GEMM:  out[M x Nout] = ReLU( concat(seg0,seg1,seg2)[M x Ktot] @ W + b )
// A rows gathered per segment: mode 0 = idx_row[g], 1 = idx_col[g], 2 = direct g.
// Each wave processes TWO 16-row M-tiles (32 rows) so every B fragment load is
// reused by two WMMAs. Weights pre-packed in B-fragment order. Optional fused
// scatter-add epilogue (atomicAdd into the aggregation buffer).
// ---------------------------------------------------------------------------
template<int NT>
__global__ __launch_bounds__(128) void wmma_gemm_gather(
    const bf16_t* __restrict__ s0, const bf16_t* __restrict__ s1,
    const bf16_t* __restrict__ s2,
    int d0, int d1, int d2, int m0, int m1, int m2,
    const int* __restrict__ idx_row, const int* __restrict__ idx_col,
    const unsigned int* __restrict__ wb, const float* __restrict__ bias,
    int Mrows, int Ktot,
    float* __restrict__ outF, bf16_t* __restrict__ outB, int relu,
    float* __restrict__ aggAtomic)
{
    const int lane  = threadIdx.x & 31;
    const int wave  = threadIdx.x >> 5;
    const int pair  = blockIdx.x * 4 + wave;     // 32-row super-tile index
    const int tbase = pair * 32;
    if (tbase >= Mrows) return;                  // whole-wave uniform exit
    const bool t1ok = (tbase + 16) < Mrows;      // second 16-row tile valid?
    const int Nout = NT * 16;
    const int half = lane >> 4;
    const int g0   = tbase + (lane & 15);        // A row for sub-tile 0
    const int g1   = t1ok ? (g0 + 16) : g0;      // clamped for safe loads

    auto segbase = [&](const bf16_t* p, int mode, int width, int g) -> const bf16_t* {
        int id = (mode == 0) ? idx_row[g] : (mode == 1) ? idx_col[g] : g;
        return p + (size_t)id * (size_t)width;
    };
    const bf16_t* p0a = segbase(s0, m0, d0, g0);
    const bf16_t* p0b = segbase(s0, m0, d0, g1);
    const bf16_t* p1a = segbase(s1, m1, d1, g0);
    const bf16_t* p1b = segbase(s1, m1, d1, g1);
    const bf16_t* p2a = d2 ? segbase(s2, m2, d2, g0) : p1a;
    const bf16_t* p2b = d2 ? segbase(s2, m2, d2, g1) : p1b;
    const int b1 = d0, b2 = d0 + d1;

    v8f acc0[NT], acc1[NT];
    #pragma unroll
    for (int nt = 0; nt < NT; ++nt)
        #pragma unroll
        for (int k = 0; k < 8; ++k) { acc0[nt][k] = 0.0f; acc1[nt][k] = 0.0f; }

    const int KT = Ktot >> 5;
    const int ko = half * 8;                     // A layout: lanes16-31 hold K+8
    for (int kt = 0; kt < KT; ++kt) {
        const int kb = kt << 5;
        const bf16_t *sa, *sb; int co;
        if      (kb < b1) { sa = p0a; sb = p0b; co = kb;      }
        else if (kb < b2) { sa = p1a; sb = p1b; co = kb - b1; }
        else              { sa = p2a; sb = p2b; co = kb - b2; }
        // A fragments: 16-bit A 16x32 layout (ISA 7.12.2); 8 u32 = two 16B runs
        v8u au0, au1;
        #pragma unroll
        for (int v = 0; v < 4; ++v) {
            au0[v] = *(const unsigned int*)(sa + co + 2 * v + ko);
            au1[v] = *(const unsigned int*)(sb + co + 2 * v + ko);
        }
        #pragma unroll
        for (int v = 4; v < 8; ++v) {
            au0[v] = *(const unsigned int*)(sa + co + 16 + 2 * (v - 4) + ko);
            au1[v] = *(const unsigned int*)(sb + co + 16 + 2 * (v - 4) + ko);
        }
        v16bf a0 = __builtin_bit_cast(v16bf, au0);
        v16bf a1 = __builtin_bit_cast(v16bf, au1);
        #pragma unroll
        for (int nt = 0; nt < NT; ++nt) {
            v8u bu = ((const v8u*)wb)[(size_t)(kt * NT + nt) * 32 + lane];
            v16bf b = __builtin_bit_cast(v16bf, bu);
            acc0[nt] = __builtin_amdgcn_wmma_f32_16x16x32_bf16(
                false, a0, false, b, (short)0, acc0[nt], false, false);
            acc1[nt] = __builtin_amdgcn_wmma_f32_16x16x32_bf16(
                false, a1, false, b, (short)0, acc1[nt], false, false);
        }
    }

    // Epilogue. C layout: lane n = lane&15 ; row m = j + 8*(lane>>4), VGPR j.
    const int n0 = lane & 15;
    if (aggAtomic) {
        int nodes0[8], nodes1[8];
        #pragma unroll
        for (int j = 0; j < 8; ++j) {
            nodes0[j] = idx_col[tbase + j + 8 * half];
            nodes1[j] = t1ok ? idx_col[tbase + 16 + j + 8 * half] : 0;
        }
        #pragma unroll
        for (int nt = 0; nt < NT; ++nt) {
            const int n = nt * 16 + n0;
            const float bv = bias[n];
            #pragma unroll
            for (int j = 0; j < 8; ++j) {
                float v0 = acc0[nt][j] + bv;
                if (relu) v0 = fmaxf(v0, 0.0f);
                atomicAdd(&aggAtomic[(size_t)nodes0[j] * Nout + n], v0);
                if (t1ok) {
                    float v1 = acc1[nt][j] + bv;
                    if (relu) v1 = fmaxf(v1, 0.0f);
                    atomicAdd(&aggAtomic[(size_t)nodes1[j] * Nout + n], v1);
                }
            }
        }
    } else {
        #pragma unroll
        for (int nt = 0; nt < NT; ++nt) {
            const int n = nt * 16 + n0;
            const float bv = bias[n];
            #pragma unroll
            for (int j = 0; j < 8; ++j) {
                float v0 = acc0[nt][j] + bv;
                if (relu) v0 = fmaxf(v0, 0.0f);
                size_t off0 = (size_t)(tbase + j + 8 * half) * Nout + n;
                if (outF) outF[off0] = v0;
                if (outB) outB[off0] = (bf16_t)v0;
                if (t1ok) {
                    float v1 = acc1[nt][j] + bv;
                    if (relu) v1 = fmaxf(v1, 0.0f);
                    size_t off1 = (size_t)(tbase + 16 + j + 8 * half) * Nout + n;
                    if (outF) outF[off1] = v1;
                    if (outB) outB[off1] = (bf16_t)v1;
                }
            }
        }
    }
}

// Pack fp32 weight [K x N] into WMMA B-fragment bf16 pairs (ISA 7.12.2 B 32x16).
__global__ void pack_weight_bf16(const float* __restrict__ W,
                                 unsigned int* __restrict__ out, int K, int N)
{
    const int NT = N >> 4;
    const int total = (K >> 5) * NT * 256;
    int i = blockIdx.x * blockDim.x + threadIdx.x;
    if (i >= total) return;
    const int v    = i & 7;
    const int lane = (i >> 3) & 31;
    const int t    = i >> 8;
    const int kt   = t / NT, nt = t - kt * NT;
    const int n    = nt * 16 + (lane & 15);
    const int k    = kt * 32 + (lane >> 4) * 8 + 2 * v + (v >= 4 ? 8 : 0);
    unsigned short lo = __builtin_bit_cast(unsigned short, (bf16_t)W[(size_t)k * N + n]);
    unsigned short hi = __builtin_bit_cast(unsigned short, (bf16_t)W[(size_t)(k + 1) * N + n]);
    out[i] = ((unsigned)hi << 16) | (unsigned)lo;
}

__global__ void zero_f32(float* __restrict__ p, long long n) {
    long long i = (long long)blockIdx.x * blockDim.x + threadIdx.x;
    if (i < n) p[i] = 0.0f;
}
__global__ void f32_to_bf16_k(const float* __restrict__ in, bf16_t* __restrict__ out, long long n) {
    long long i = (long long)blockIdx.x * blockDim.x + threadIdx.x;
    if (i < n) out[i] = (bf16_t)in[i];
}
__global__ void count_edges(const int* __restrict__ col, float* __restrict__ cnt, int E) {
    int i = blockIdx.x * blockDim.x + threadIdx.x;
    if (i < E) atomicAdd(&cnt[col[i]], 1.0f);
}
__global__ void agg_norm_bf16(const float* __restrict__ agg, const float* __restrict__ cnt,
                              bf16_t* __restrict__ out, int Nn, int C) {
    long long i = (long long)blockIdx.x * blockDim.x + threadIdx.x;
    if (i >= (long long)Nn * C) return;
    int node = (int)(i / C);
    out[i] = (bf16_t)(agg[i] / fmaxf(cnt[node], 1.0f));
}
// per-column sum / sumsq;  blockDim.x == C
__global__ void bn_stats(const float* __restrict__ X, int M, int C,
                         float* __restrict__ sum, float* __restrict__ sq) {
    const int c = threadIdx.x;
    const int r0 = blockIdx.x * 256;
    const int rend = min(r0 + 256, M);
    float s = 0.0f, q = 0.0f;
    for (int r = r0; r < rend; ++r) {
        float v = X[(size_t)r * C + c];
        s += v; q += v * v;
    }
    atomicAdd(&sum[c], s);
    atomicAdd(&sq[c], q);
}
__global__ void bn_apply_bf16(const float* __restrict__ X, bf16_t* __restrict__ out,
                              long long M, int C,
                              const float* __restrict__ sum, const float* __restrict__ sq,
                              const float* __restrict__ gamma, const float* __restrict__ beta) {
    long long i = (long long)blockIdx.x * blockDim.x + threadIdx.x;
    if (i >= M * C) return;
    int c = (int)(i % C);
    float mu  = sum[c] / (float)M;
    float var = sq[c] / (float)M - mu * mu;
    out[i] = (bf16_t)((X[i] - mu) * rsqrtf(var + EPS) * gamma[c] + beta[c]);
}

// ---------------------------------------------------------------------------
static inline char* carve(char*& p, size_t bytes) {
    char* r = p;
    p += (bytes + 255) & ~(size_t)255;
    return r;
}

static void run_gemm(int Nout, int Mrows,
                     const bf16_t* s0, const bf16_t* s1, const bf16_t* s2,
                     int d0, int d1, int d2, int m0, int m1, int m2,
                     const int* row, const int* col,
                     const unsigned* wb, const float* bias, int Ktot,
                     float* outF, bf16_t* outB, int relu, float* aggA,
                     hipStream_t stream)
{
    const int pairs  = (Mrows + 31) / 32;        // 32-row super-tiles
    const int blocks = (pairs + 3) / 4;          // 4 waves per block
    if (Nout == 128)
        wmma_gemm_gather<8><<<blocks, 128, 0, stream>>>(s0, s1, s2, d0, d1, d2, m0, m1, m2,
            row, col, wb, bias, Mrows, Ktot, outF, outB, relu, aggA);
    else
        wmma_gemm_gather<4><<<blocks, 128, 0, stream>>>(s0, s1, s2, d0, d1, d2, m0, m1, m2,
            row, col, wb, bias, Mrows, Ktot, outF, outB, relu, aggA);
}

struct LayerP { const float *ew,*eb,*n1w,*n1b,*n2w,*n2b,*ng,*nb,*eg,*ebt; };

extern "C" void kernel_launch(void* const* d_in, const int* in_sizes, int n_in,
                              void* d_out, int out_size, void* d_ws, size_t ws_size,
                              hipStream_t stream)
{
    // ---- input mapping: detect tree-sorted vs insertion-order flattening ----
    LayerP par[4] = {};
    const float* x_in = nullptr;
    const float* e_attr = nullptr;
    const int*   eidx = nullptr;
    const bool sorted = (in_sizes[0] == NE * 64);   // edge_attr first => tree-sorted
    if (sorted) {
        e_attr = (const float*)d_in[0];
        eidx   = (const int*)d_in[1];
        int ix = 2;
        for (int l = 0; l < 4; ++l) {
            if (l < 3) { // keys: eb, edge_beta, edge_gamma, ew, n1b, n1w, n2b, n2w, node_beta, node_gamma
                par[l].eb  = (const float*)d_in[ix + 0];
                par[l].ebt = (const float*)d_in[ix + 1];
                par[l].eg  = (const float*)d_in[ix + 2];
                par[l].ew  = (const float*)d_in[ix + 3];
                par[l].n1b = (const float*)d_in[ix + 4];
                par[l].n1w = (const float*)d_in[ix + 5];
                par[l].n2b = (const float*)d_in[ix + 6];
                par[l].n2w = (const float*)d_in[ix + 7];
                par[l].nb  = (const float*)d_in[ix + 8];
                par[l].ng  = (const float*)d_in[ix + 9];
                ix += 10;
            } else {     // keys: eb, ew, n1b, n1w, n2b, n2w
                par[l].eb  = (const float*)d_in[ix + 0];
                par[l].ew  = (const float*)d_in[ix + 1];
                par[l].n1b = (const float*)d_in[ix + 2];
                par[l].n1w = (const float*)d_in[ix + 3];
                par[l].n2b = (const float*)d_in[ix + 4];
                par[l].n2w = (const float*)d_in[ix + 5];
                ix += 6;
            }
        }
        x_in = (const float*)d_in[ix];
    } else {             // insertion order: x, edge_index, edge_attr, params
        x_in   = (const float*)d_in[0];
        eidx   = (const int*)d_in[1];
        e_attr = (const float*)d_in[2];
        int ix = 3;
        for (int l = 0; l < 4; ++l) {
            par[l].ew  = (const float*)d_in[ix + 0];
            par[l].eb  = (const float*)d_in[ix + 1];
            par[l].n1w = (const float*)d_in[ix + 2];
            par[l].n1b = (const float*)d_in[ix + 3];
            par[l].n2w = (const float*)d_in[ix + 4];
            par[l].n2b = (const float*)d_in[ix + 5];
            if (l < 3) {
                par[l].ng  = (const float*)d_in[ix + 6];
                par[l].nb  = (const float*)d_in[ix + 7];
                par[l].eg  = (const float*)d_in[ix + 8];
                par[l].ebt = (const float*)d_in[ix + 9];
                ix += 10;
            } else ix += 6;
        }
    }
    const int* row = eidx;
    const int* col = eidx + NE;

    // ---- workspace carve ----
    char* wp = (char*)d_ws;
    bf16_t*   xb   = (bf16_t*)carve(wp, (size_t)NN * 128 * 2);
    bf16_t*   ebA  = (bf16_t*)carve(wp, (size_t)NE * 128 * 2);
    bf16_t*   ebB  = (bf16_t*)carve(wp, (size_t)NE * 128 * 2);
    float*    eF   = (float*) carve(wp, (size_t)NE * 128 * 4);
    float*    hF   = (float*) carve(wp, (size_t)NN * 128 * 4);
    float*    agg  = (float*) carve(wp, (size_t)NN * 128 * 4);
    bf16_t*   aggb = (bf16_t*)carve(wp, (size_t)NN * 128 * 2);
    float*    cnt  = (float*) carve(wp, (size_t)NN * 4);
    float*    st   = (float*) carve(wp, 4 * 128 * 4);   // sumN, sqN, sumE, sqE
    unsigned* wbE  = (unsigned*)carve(wp, 24576 * 4);
    unsigned* wbN1 = (unsigned*)carve(wp, 24576 * 4);
    unsigned* wbN2 = (unsigned*)carve(wp, 24576 * 4);

    auto g1 = [](long long n) { return (unsigned)((n + 255) / 256); };

    // ---- initial bf16 conversions + edge counts ----
    f32_to_bf16_k<<<g1((long long)NN * 64), 256, 0, stream>>>(x_in, xb, (long long)NN * 64);
    f32_to_bf16_k<<<g1((long long)NE * 64), 256, 0, stream>>>(e_attr, ebA, (long long)NE * 64);
    zero_f32<<<g1(NN), 256, 0, stream>>>(cnt, NN);
    count_edges<<<g1(NE), 256, 0, stream>>>(col, cnt, NE);

    // ---- layer loop ----
    const int dims[4][4] = {{64,64,128,128},{128,128,128,128},{128,128,128,128},{128,128,64,64}};
    bf16_t* ebIn = ebA;
    bf16_t* ebOut = ebB;
    float* out_nodes = (float*)d_out;                    // 50000 x 64
    float* out_edges = (float*)d_out + (size_t)NN * 64;  // 400000 x 64

    for (int l = 0; l < 4; ++l) {
        const int dn = dims[l][0], de = dims[l][1], eout = dims[l][2], nout = dims[l][3];
        const bool last = (l == 3);
        const int Ke = 2 * dn + de, Kn1 = dn + eout, Kn2 = dn + nout;

        // pack weights into WMMA fragment order
        pack_weight_bf16<<<g1((Ke >> 5) * (eout >> 4) * 256), 256, 0, stream>>>(par[l].ew,  wbE,  Ke,  eout);
        pack_weight_bf16<<<g1((Kn1 >> 5) * (nout >> 4) * 256), 256, 0, stream>>>(par[l].n1w, wbN1, Kn1, nout);
        pack_weight_bf16<<<g1((Kn2 >> 5) * (nout >> 4) * 256), 256, 0, stream>>>(par[l].n2w, wbN2, Kn2, nout);

        // EdgeModel: e = [x[row] | x[col] | edge_attr] @ ew + eb  (+ReLU)
        float* eOutF = last ? out_edges : eF;
        run_gemm(eout, NE, xb, xb, ebIn, dn, dn, de, 0, 1, 2, row, col,
                 wbE, par[l].eb, Ke, eOutF, ebOut, !last, nullptr, stream);

        // NodeModel mlp1 + fused scatter-add: m = [x[row] | e] @ n1w + n1b (+ReLU)
        zero_f32<<<g1((long long)NN * nout), 256, 0, stream>>>(agg, (long long)NN * nout);
        run_gemm(nout, NE, xb, ebOut, ebOut, dn, eout, 0, 0, 2, 2, row, col,
                 wbN1, par[l].n1b, Kn1, nullptr, nullptr, !last, agg, stream);

        // scatter-mean normalize -> bf16
        agg_norm_bf16<<<g1((long long)NN * nout), 256, 0, stream>>>(agg, cnt, aggb, NN, nout);

        // NodeModel mlp2: h = [x | agg] @ n2w + n2b (+ReLU)
        float* hOutF = last ? out_nodes : hF;
        run_gemm(nout, NN, xb, aggb, aggb, dn, nout, 0, 2, 2, 2, row, col,
                 wbN2, par[l].n2b, Kn2, hOutF, nullptr, !last, nullptr, stream);

        if (!last) {
            // BatchNorm (training mode, biased var) + bf16 convert for next layer
            zero_f32<<<2, 256, 0, stream>>>(st, 512);
            bn_stats<<<(NN + 255) / 256, nout, 0, stream>>>(hF, NN, nout, st, st + 128);
            bn_apply_bf16<<<g1((long long)NN * nout), 256, 0, stream>>>(
                hF, xb, NN, nout, st, st + 128, par[l].ng, par[l].nb);
            bn_stats<<<(NE + 255) / 256, eout, 0, stream>>>(eF, NE, eout, st + 256, st + 384);
            bn_apply_bf16<<<g1((long long)NE * eout), 256, 0, stream>>>(
                eF, ebOut, NE, eout, st + 256, st + 384, par[l].eg, par[l].ebt);
            // ping-pong edge buffers
            bf16_t* t = ebIn; ebIn = ebOut; ebOut = t;
        }
    }
}
